// TransformerDecoderLayer_49392123904487
// MI455X (gfx1250) — matrix operand order repaired and checked by reading, not compiled, not run on hardware
//
#include <hip/hip_runtime.h>

typedef __attribute__((ext_vector_type(16))) __bf16 v16bf;
typedef __attribute__((ext_vector_type(8)))  float  v8f;

union V16U { v16bf bf; uint4 q[2]; };

__device__ __forceinline__ unsigned short f2bf(float f) {
    unsigned u = __float_as_uint(f);
    unsigned r = u + 0x7FFFu + ((u >> 16) & 1u);   // round-to-nearest-even
    return (unsigned short)(r >> 16);
}

// ---------------------------------------------------------------------------
// Prep kernels
// ---------------------------------------------------------------------------

// zero-fill n uint4s (16B each)
__global__ void zero_kernel(uint4* __restrict__ p, int n) {
    int tid = blockIdx.x * blockDim.x + threadIdx.x;
    if (tid < n) p[tid] = make_uint4(0, 0, 0, 0);
}

// (C,N) f32 -> (N,C) bf16
__global__ void tconv_kernel(const float* __restrict__ in,
                             unsigned short* __restrict__ out, int C, int N) {
    int tid = blockIdx.x * blockDim.x + threadIdx.x;
    if (tid >= C * N) return;
    int n = tid / C, c = tid % C;
    out[tid] = f2bf(in[(size_t)c * N + n]);
}

// elementwise f32 -> bf16
__global__ void cvt_kernel(const float* __restrict__ in,
                           unsigned short* __restrict__ out, int n) {
    int tid = blockIdx.x * blockDim.x + threadIdx.x;
    if (tid < n) out[tid] = f2bf(in[tid]);
}

// (O,C,3,3) f32 -> (9,O,C) bf16  (per-tap weight matrices, K-contiguous)
__global__ void w3prep_kernel(const float* __restrict__ in,
                              unsigned short* __restrict__ out, int OC) {
    int tid = blockIdx.x * blockDim.x + threadIdx.x;
    if (tid >= 9 * OC) return;
    int t = tid / OC, oc = tid % OC;
    out[tid] = f2bf(in[(size_t)oc * 9 + t]);
}

// ---------------------------------------------------------------------------
// Generic implicit-GEMM conv: dst(O,N) = sum_t W_t(O,IC) * shift_t(src)(IC,N)
// wt: (taps,O,IC) bf16, src: (N,IC) bf16 pixel-major, one wave = 16x16 tile.
// OOB pixels read from a zero-filled row (per-lane pointer select, no EXEC
// masking around the loads, keeps the WMMA loop branch-free).
// ---------------------------------------------------------------------------
__global__ void __launch_bounds__(256)
convgemm_kernel(const unsigned short* __restrict__ wt,
                const unsigned short* __restrict__ src,
                const unsigned short* __restrict__ zrow,   // >= IC zero bf16
                float* __restrict__ dst,
                const float* __restrict__ bias,            // never null
                int O, int IC, int N, int Hh, int Ww, int taps, int dil) {
    const int lane = threadIdx.x & 31;
    const int wave = threadIdx.x >> 5;
    const int col  = lane & 15;
    const int hl   = lane >> 4;
    const int otile = blockIdx.y * 8 + wave;
    const int pix   = blockIdx.x * 16 + col;
    const int hh = pix / Ww, ww = pix % Ww;

    v8f acc = {};
    const unsigned short* wrow = wt + (size_t)(otile * 16 + col) * IC;
    for (int t = 0; t < taps; ++t) {
        int dh = (taps == 1) ? 0 : ((t / 3) - 1) * dil;
        int dw = (taps == 1) ? 0 : ((t % 3) - 1) * dil;
        int sh = hh + dh, sw = ww + dw;
        bool valid = (sh >= 0) && (sh < Hh) && (sw >= 0) && (sw < Ww);
        const unsigned short* srow =
            valid ? (src + (size_t)(sh * Ww + sw) * IC) : zrow;
        const unsigned short* wr = wrow + (size_t)t * O * IC;
        for (int c0 = 0; c0 < IC; c0 += 32) {
            V16U a, b;
            a.q[0] = *(const uint4*)(wr + c0 + hl * 8);
            a.q[1] = *(const uint4*)(wr + c0 + 16 + hl * 8);
            b.q[0] = *(const uint4*)(srow + c0 + hl * 8);
            b.q[1] = *(const uint4*)(srow + c0 + 16 + hl * 8);
            acc = __builtin_amdgcn_wmma_f32_16x16x32_bf16(
                false, a.bf, false, b.bf, (short)0, acc, false, false);
        }
    }
#pragma unroll
    for (int r = 0; r < 8; ++r) {
        int o = otile * 16 + r + 8 * hl;
        dst[(size_t)o * N + pix] = acc[r] + bias[o];
    }
}

// ---------------------------------------------------------------------------
// Q: L2-normalize over D, write (h,N,D) bf16
// ---------------------------------------------------------------------------
__global__ void qnorm_kernel(const float* __restrict__ Qraw,
                             unsigned short* __restrict__ Qn, int N) {
    int tid = blockIdx.x * blockDim.x + threadIdx.x;
    if (tid >= 8 * N) return;
    int h = tid / N, n = tid % N;
    float vals[32];
    float s = 0.0f;
#pragma unroll
    for (int d = 0; d < 32; ++d) {
        float v = Qraw[(size_t)(h * 32 + d) * N + n];
        vals[d] = v; s += v * v;
    }
    float inv = 1.0f / fmaxf(sqrtf(s), 1e-12f);
#pragma unroll
    for (int d = 0; d < 32; ++d)
        Qn[((size_t)h * N + n) * 32 + d] = f2bf(vals[d] * inv);
}

// ---------------------------------------------------------------------------
// K: L2-normalize over the HEADS axis (faithful to ref), write (h,N,D) bf16
// ---------------------------------------------------------------------------
__global__ void knorm_kernel(const float* __restrict__ Kraw,
                             unsigned short* __restrict__ Knt, int N) {
    int tid = blockIdx.x * blockDim.x + threadIdx.x;
    if (tid >= 32 * N) return;
    int d = tid / N, n = tid % N;
    float vals[8];
    float s = 0.0f;
#pragma unroll
    for (int h = 0; h < 8; ++h) {
        float v = Kraw[(size_t)(h * 32 + d) * N + n];
        vals[h] = v; s += v * v;
    }
    float inv = 1.0f / fmaxf(sqrtf(s), 1e-12f);
#pragma unroll
    for (int h = 0; h < 8; ++h)
        Knt[((size_t)h * N + n) * 32 + d] = f2bf(vals[h] * inv);
}

// ---------------------------------------------------------------------------
// Flash attention: one wave = 16 query rows; stream K/V in 32-col chunks.
// Qn (h,N,32) bf16, K (h,N,32) bf16 (pre-normalized, m-major), Vt (h,32,N)
// bf16. Writes cat (N, 768) bf16, channels [z*256 + h*32 .. +31].
// ---------------------------------------------------------------------------
__global__ void __launch_bounds__(128)
attn_kernel(const unsigned short* __restrict__ Qn,
            const unsigned short* __restrict__ K1,
            const unsigned short* __restrict__ K2,
            const unsigned short* __restrict__ K3,
            const unsigned short* __restrict__ Vt,
            unsigned short* __restrict__ cat, int N) {
    __shared__ __align__(16) unsigned short ldsP[4][16 * 32];

    const int lane = threadIdx.x & 31;
    const int wave = threadIdx.x >> 5;
    const int col  = lane & 15;
    const int hl   = lane >> 4;
    const int h = blockIdx.y;
    const int z = blockIdx.z;
    const unsigned short* K = (z == 0) ? K1 : ((z == 1) ? K2 : K3);
    const int n0 = (blockIdx.x * 4 + wave) * 16;

    // Q A-operand, loaded once (row = n0+col, K = d pattern)
    const unsigned short* qrow = Qn + ((size_t)h * N + (n0 + col)) * 32;
    V16U aq;
    aq.q[0] = *(const uint4*)(qrow + hl * 8);
    aq.q[1] = *(const uint4*)(qrow + 16 + hl * 8);

    v8f O0 = {}, O1 = {};
    float m_r[8], l_r[8];
#pragma unroll
    for (int r = 0; r < 8; ++r) { m_r[r] = -1e30f; l_r[r] = 0.0f; }

    const unsigned short* Kh = K  + (size_t)h * N * 32;
    const unsigned short* Vh = Vt + (size_t)h * 32 * N;
    unsigned short* myP = ldsP[wave];

    for (int m0 = 0; m0 < N; m0 += 32) {
        // ---- S = Qn * Kn^T  (two 16x16 tiles, K-dim = D = 32) ----
        V16U b0, b1;
        const unsigned short* kr0 = Kh + (size_t)(m0 + col) * 32 + hl * 8;
        const unsigned short* kr1 = Kh + (size_t)(m0 + 16 + col) * 32 + hl * 8;
        b0.q[0] = *(const uint4*)(kr0); b0.q[1] = *(const uint4*)(kr0 + 16);
        b1.q[0] = *(const uint4*)(kr1); b1.q[1] = *(const uint4*)(kr1 + 16);

        // prefetch next iteration's K/V cache lines (global_prefetch_b8)
        if (m0 + 32 < N) {
            __builtin_prefetch(kr0 + 32 * 32, 0, 3);
            __builtin_prefetch(kr1 + 32 * 32, 0, 3);
            __builtin_prefetch(Vh + (size_t)col * N + m0 + 32 + hl * 16, 0, 3);
            __builtin_prefetch(Vh + (size_t)(16 + col) * N + m0 + 32 + hl * 16, 0, 3);
        }

        v8f S0 = {}, S1 = {};
        S0 = __builtin_amdgcn_wmma_f32_16x16x32_bf16(
            false, aq.bf, false, b0.bf, (short)0, S0, false, false);
        S1 = __builtin_amdgcn_wmma_f32_16x16x32_bf16(
            false, aq.bf, false, b1.bf, (short)0, S1, false, false);

        // ---- online softmax: C-layout row r+8*hl lives in VGPR r ----
#pragma unroll
        for (int r = 0; r < 8; ++r) {
            float t = fmaxf(S0[r], S1[r]);
            t = fmaxf(t, __shfl_xor(t, 1, 16));
            t = fmaxf(t, __shfl_xor(t, 2, 16));
            t = fmaxf(t, __shfl_xor(t, 4, 16));
            t = fmaxf(t, __shfl_xor(t, 8, 16));
            float mnew  = fmaxf(m_r[r], t);
            float alpha = __expf(m_r[r] - mnew);
            float p0 = __expf(S0[r] - mnew);
            float p1 = __expf(S1[r] - mnew);
            S0[r] = p0; S1[r] = p1;
            float s = p0 + p1;
            s += __shfl_xor(s, 1, 16);
            s += __shfl_xor(s, 2, 16);
            s += __shfl_xor(s, 4, 16);
            s += __shfl_xor(s, 8, 16);
            l_r[r] = l_r[r] * alpha + s;
            m_r[r] = mnew;
            O0[r] *= alpha;
            O1[r] *= alpha;
        }

        // ---- C-layout -> A-layout transpose of P through LDS ----
#pragma unroll
        for (int r = 0; r < 8; ++r) {
            int row = r + 8 * hl;
            myP[row * 32 + col]      = f2bf(S0[r]);
            myP[row * 32 + 16 + col] = f2bf(S1[r]);
        }
        V16U ap;
        const unsigned short* pr = myP + col * 32;
        ap.q[0] = *(const uint4*)(pr + hl * 8);
        ap.q[1] = *(const uint4*)(pr + 16 + hl * 8);

        // ---- O += P * V  (K-dim = m chunk of 32) ----
        V16U vb0, vb1;
        const unsigned short* vr0 = Vh + (size_t)col * N + m0 + hl * 8;
        const unsigned short* vr1 = Vh + (size_t)(16 + col) * N + m0 + hl * 8;
        vb0.q[0] = *(const uint4*)(vr0); vb0.q[1] = *(const uint4*)(vr0 + 16);
        vb1.q[0] = *(const uint4*)(vr1); vb1.q[1] = *(const uint4*)(vr1 + 16);
        O0 = __builtin_amdgcn_wmma_f32_16x16x32_bf16(
            false, ap.bf, false, vb0.bf, (short)0, O0, false, false);
        O1 = __builtin_amdgcn_wmma_f32_16x16x32_bf16(
            false, ap.bf, false, vb1.bf, (short)0, O1, false, false);
    }

    // ---- normalize + write (N, 768) bf16 ----
#pragma unroll
    for (int r = 0; r < 8; ++r) {
        int n = n0 + r + 8 * hl;
        float inv = 1.0f / l_r[r];
        size_t base = (size_t)n * 768 + (size_t)z * 256 + h * 32;
        cat[base + col]      = f2bf(O0[r] * inv);
        cat[base + 16 + col] = f2bf(O1[r] * inv);
    }
}

// ---------------------------------------------------------------------------
// Host-side launch
// ---------------------------------------------------------------------------
extern "C" void kernel_launch(void* const* d_in, const int* in_sizes, int n_in,
                              void* d_out, int out_size, void* d_ws, size_t ws_size,
                              hipStream_t stream) {
    const int C = 256, H = 56, W = 56, N = H * W, NH = 8, IC3 = 768;

    const float* x   = (const float*)d_in[0];
    const float* y   = (const float*)d_in[1];
    const float* wq  = (const float*)d_in[2];
    const float* bq  = (const float*)d_in[3];
    const float* wk1 = (const float*)d_in[4];
    const float* wk2 = (const float*)d_in[5];
    const float* wk3 = (const float*)d_in[6];
    const float* wv  = (const float*)d_in[7];
    const float* bv  = (const float*)d_in[8];
    const float* wo  = (const float*)d_in[9];
    const float* bo  = (const float*)d_in[10];
    float* out = (float*)d_out;

    char* ws = (char*)d_ws;
    size_t off = 0;
    auto alloc = [&](size_t bytes) -> void* {
        void* p = ws + off;
        off += (bytes + 255) & ~size_t(255);
        return p;
    };

    // zero scratch: bf16 zero row (>= IC3*2 = 1536B) + f32 zero bias (1024B)
    char* zbase = (char*)alloc(8192);
    unsigned short* zrow  = (unsigned short*)zbase;          // 1536B used
    float*          zbias = (float*)(zbase + 4096);          // 1024B used

    unsigned short* xt   = (unsigned short*)alloc((size_t)N * C * 2);
    unsigned short* yt   = (unsigned short*)alloc((size_t)N * C * 2);
    unsigned short* wqt  = (unsigned short*)alloc((size_t)C * C * 2);
    unsigned short* wvt  = (unsigned short*)alloc((size_t)C * C * 2);
    unsigned short* wot  = (unsigned short*)alloc((size_t)C * IC3 * 2);
    unsigned short* wk1t = (unsigned short*)alloc((size_t)9 * C * C * 2);
    unsigned short* wk2t = (unsigned short*)alloc((size_t)9 * C * C * 2);
    unsigned short* wk3t = (unsigned short*)alloc((size_t)9 * C * C * 2);
    float* Qraw  = (float*)alloc((size_t)C * N * 4);
    float* Vraw  = (float*)alloc((size_t)C * N * 4);
    float* K1raw = (float*)alloc((size_t)C * N * 4);
    float* K2raw = (float*)alloc((size_t)C * N * 4);
    float* K3raw = (float*)alloc((size_t)C * N * 4);
    unsigned short* Qn  = (unsigned short*)alloc((size_t)NH * N * 32 * 2);
    unsigned short* Kn1 = (unsigned short*)alloc((size_t)NH * N * 32 * 2);
    unsigned short* Kn2 = (unsigned short*)alloc((size_t)NH * N * 32 * 2);
    unsigned short* Kn3 = (unsigned short*)alloc((size_t)NH * N * 32 * 2);
    unsigned short* Vb  = (unsigned short*)alloc((size_t)C * N * 2);
    unsigned short* cat = (unsigned short*)alloc((size_t)N * IC3 * 2);

    // ---- preps ----
    zero_kernel<<<2, 256, 0, stream>>>((uint4*)zbase, 8192 / 16);
    tconv_kernel<<<(C * N + 255) / 256, 256, 0, stream>>>(x, xt, C, N);
    tconv_kernel<<<(C * N + 255) / 256, 256, 0, stream>>>(y, yt, C, N);
    cvt_kernel<<<(C * C + 255) / 256, 256, 0, stream>>>(wq, wqt, C * C);
    cvt_kernel<<<(C * C + 255) / 256, 256, 0, stream>>>(wv, wvt, C * C);
    cvt_kernel<<<(C * IC3 + 255) / 256, 256, 0, stream>>>(wo, wot, C * IC3);
    w3prep_kernel<<<(9 * C * C + 255) / 256, 256, 0, stream>>>(wk1, wk1t, C * C);
    w3prep_kernel<<<(9 * C * C + 255) / 256, 256, 0, stream>>>(wk2, wk2t, C * C);
    w3prep_kernel<<<(9 * C * C + 255) / 256, 256, 0, stream>>>(wk3, wk3t, C * C);

    // ---- convs (implicit GEMM, WMMA) ----
    dim3 cg(N / 16, 2);
    convgemm_kernel<<<cg, 256, 0, stream>>>(wqt,  xt, zrow, Qraw,  bq,    C, C,   N, H, W, 1, 1);
    convgemm_kernel<<<cg, 256, 0, stream>>>(wvt,  yt, zrow, Vraw,  bv,    C, C,   N, H, W, 1, 1);
    convgemm_kernel<<<cg, 256, 0, stream>>>(wk1t, yt, zrow, K1raw, zbias, C, C,   N, H, W, 9, 1);
    convgemm_kernel<<<cg, 256, 0, stream>>>(wk2t, yt, zrow, K2raw, zbias, C, C,   N, H, W, 9, 2);
    convgemm_kernel<<<cg, 256, 0, stream>>>(wk3t, yt, zrow, K3raw, zbias, C, C,   N, H, W, 9, 3);

    // ---- normalizations / layout conversions ----
    qnorm_kernel<<<(NH * N + 255) / 256, 256, 0, stream>>>(Qraw, Qn, N);
    knorm_kernel<<<(32 * N + 255) / 256, 256, 0, stream>>>(K1raw, Kn1, N);
    knorm_kernel<<<(32 * N + 255) / 256, 256, 0, stream>>>(K2raw, Kn2, N);
    knorm_kernel<<<(32 * N + 255) / 256, 256, 0, stream>>>(K3raw, Kn3, N);
    cvt_kernel<<<(C * N + 255) / 256, 256, 0, stream>>>(Vraw, Vb, C * N);

    // ---- flash attention, 3 branches x 8 heads ----
    attn_kernel<<<dim3(N / 64, NH, 3), 128, 0, stream>>>(Qn, Kn1, Kn2, Kn3, Vb, cat, N);

    // ---- final projection (C x 3C) * (3C x N) + bias ----
    convgemm_kernel<<<cg, 256, 0, stream>>>(wot, cat, zrow, out, bo, C, IC3, N, H, W, 1, 1);
}
